// OptimizedCrossModalAttention_49752901157248
// MI455X (gfx1250) — matrix operand order repaired and checked by reading, not compiled, not run on hardware
//
#include <hip/hip_runtime.h>

// ---------------------------------------------------------------------------
// Types for CDNA5 WMMA (wave32)
// ---------------------------------------------------------------------------
typedef __attribute__((ext_vector_type(16))) __bf16          v16bf;
typedef __attribute__((ext_vector_type(8)))  float           v8f;
typedef __attribute__((ext_vector_type(8)))  unsigned short  u16x8;   // 16B
typedef __attribute__((ext_vector_type(16))) unsigned short  u16x16;  // 32B

union V16 {
    v16bf          v;
    u16x8          h[2];
    unsigned short s[16];
};

__device__ __forceinline__ unsigned short f2bf(float f) {
    unsigned int u = __builtin_bit_cast(unsigned int, f);
    unsigned int r = u + 0x7fffu + ((u >> 16) & 1u);   // round-to-nearest-even
    return (unsigned short)(r >> 16);
}

__device__ __forceinline__ void cvt4(float4 f, unsigned short* dst) {
    dst[0] = f2bf(f.x); dst[1] = f2bf(f.y); dst[2] = f2bf(f.z); dst[3] = f2bf(f.w);
}

__device__ __forceinline__ v8f wmma_bf16(v16bf a, v16bf b, v8f c) {
    // (neg_a, A, neg_b, B, c_mod, C, reuse_a, reuse_b)
    return __builtin_amdgcn_wmma_f32_16x16x32_bf16(false, a, false, b, (short)0, c, false, false);
}

__device__ __forceinline__ v16bf ldB32(const unsigned short* p) {
    u16x16 raw = *(const u16x16*)p;
    return __builtin_bit_cast(v16bf, raw);
}

// ---------------------------------------------------------------------------
// Stage 0: transpose weights fp32 -> bf16, N-major (Wt[n*K + k] = W[k*N + n])
// ---------------------------------------------------------------------------
__global__ void transpose_bf16_kernel(const float* __restrict__ W,
                                      unsigned short* __restrict__ Wt,
                                      int K, int N) {
    int i = blockIdx.x * blockDim.x + threadIdx.x;
    if (i >= K * N) return;
    int n = i / K;
    int k = i - n * K;
    Wt[i] = f2bf(W[k * N + n]);
}

// ---------------------------------------------------------------------------
// Stage 1: QKV projection. Y[4096,3072] = X[4096,1024] @ Wqkv, per modality.
// Epilogue scatters into head-separated Q,K (row-major [bh][S][64]) and
// transposed V ([bh][64][S]) as bf16.
// Wave tile: 16(M) x 128(N), 8 waves/block.
// ---------------------------------------------------------------------------
__global__ __launch_bounds__(256)
void qkv_gemm_kernel(const float* __restrict__ Xv, const float* __restrict__ Xt,
                     const unsigned short* __restrict__ WT,     // [3072][1024] bf16
                     unsigned short* __restrict__ Qb,
                     unsigned short* __restrict__ Kb,
                     unsigned short* __restrict__ Vt) {
    const int lane = threadIdx.x & 31;
    const int wave = threadIdx.x >> 5;
    const int lo   = lane & 15;
    const int hi   = lane >> 4;
    const int mod  = blockIdx.z;
    const float* __restrict__ X = mod ? Xt : Xv;

    const int m_tile = (blockIdx.y * 8 + wave) * 16;
    const int n_base = blockIdx.x * 128;

    v8f acc[8];
    for (int i = 0; i < 8; ++i)
        for (int j = 0; j < 8; ++j) acc[i][j] = 0.0f;

    const float* __restrict__ arow = X + (size_t)(m_tile + lo) * 1024;

    for (int k0 = 0; k0 < 1024; k0 += 32) {
        __builtin_prefetch(arow + k0 + 64, 0, 0);  // global_prefetch_b8
        V16 a;
        const float4* pa = (const float4*)(arow + k0 + hi * 16);
        cvt4(pa[0], &a.s[0]);  cvt4(pa[1], &a.s[4]);
        const float4* pb = (const float4*)(arow + k0 + hi * 16 + 8);
        cvt4(pb[0], &a.s[8]);  cvt4(pb[1], &a.s[12]);

        for (int nt = 0; nt < 8; ++nt) {
            const int n = n_base + nt * 16 + lo;
            v16bf b = ldB32(WT + (size_t)n * 1024 + k0 + hi * 16);
            acc[nt] = wmma_bf16(a.v, b, acc[nt]);
        }
    }

    // Epilogue: scatter C fragments (lane: col = lo, rows = v + 8*hi)
    for (int nt = 0; nt < 8; ++nt) {
        const int n   = n_base + nt * 16 + lo;
        const int sec = n >> 10;          // 0=Q 1=K 2=V
        const int h   = (n >> 6) & 15;
        const int d   = n & 63;
        for (int v = 0; v < 8; ++v) {
            const int m  = m_tile + v + 8 * hi;
            const int bb = m >> 11;
            const int s  = m & 2047;
            const unsigned short val = f2bf(acc[nt][v]);
            const int bh = (mod * 2 + bb) * 16 + h;
            if (sec == 0)      Qb[(((size_t)bh * 2048 + s) << 6) + d] = val;
            else if (sec == 1) Kb[(((size_t)bh * 2048 + s) << 6) + d] = val;
            else               Vt[(((size_t)bh * 64 + d) << 11) + s]  = val;
        }
    }
}

// ---------------------------------------------------------------------------
// Stage 2: flash attention. Per wave: 16 q-rows, stream keys in chunks of 32.
// dir 0: Q=visual, K/V=transaction -> Attn0 ; dir 1: swapped -> Attn1.
// ---------------------------------------------------------------------------
__global__ __launch_bounds__(128)
void attention_kernel(const unsigned short* __restrict__ Qb,
                      const unsigned short* __restrict__ Kb,
                      const unsigned short* __restrict__ Vt,
                      float* __restrict__ Attn0, float* __restrict__ Attn1) {
    __shared__ unsigned short Plds[4][16 * 32];

    const int lane = threadIdx.x & 31;
    const int wave = threadIdx.x >> 5;
    const int lo   = lane & 15;
    const int hi   = lane >> 4;

    const int dir = blockIdx.z;
    const int bh  = blockIdx.y;           // b*16 + h
    const int b   = bh >> 4;
    const int h   = bh & 15;
    const int q_base = (blockIdx.x * 4 + wave) * 16;

    const int qmod  = dir;
    const int kvmod = 1 - dir;
    const unsigned short* __restrict__ Qp = Qb + ((size_t)((qmod * 2 + b) * 16 + h)) * 2048 * 64;
    const unsigned short* __restrict__ Kp = Kb + ((size_t)((kvmod * 2 + b) * 16 + h)) * 2048 * 64;
    const unsigned short* __restrict__ Vp = Vt + ((size_t)((kvmod * 2 + b) * 16 + h)) * 64 * 2048;
    float* __restrict__ Out = dir ? Attn1 : Attn0;

    // Q resident as two A-fragments (hd = 64 = 2 x K32)
    v16bf qf[2];
    {
        const unsigned short* qrow = Qp + (size_t)(q_base + lo) * 64;
        for (int ks = 0; ks < 2; ++ks) {
            V16 a;
            a.h[0] = *(const u16x8*)(qrow + ks * 32 + hi * 16);
            a.h[1] = *(const u16x8*)(qrow + ks * 32 + hi * 16 + 8);
            qf[ks] = a.v;
        }
    }

    v8f o[4];
    for (int i = 0; i < 4; ++i)
        for (int j = 0; j < 8; ++j) o[i][j] = 0.0f;
    float mrow[8], lrow[8];
    for (int v = 0; v < 8; ++v) { mrow[v] = -3.0e38f; lrow[v] = 0.0f; }

    const float scale = 0.125f;   // (64)^-0.5
    unsigned short* Pw = Plds[wave];

    for (int kc = 0; kc < 2048; kc += 32) {
        // ---- S = Q @ K^T  (two 16x16 key tiles)
        v8f s0, s1;
        for (int j = 0; j < 8; ++j) { s0[j] = 0.0f; s1[j] = 0.0f; }
        for (int ks = 0; ks < 2; ++ks) {
            v16bf b0 = ldB32(Kp + (size_t)(kc + lo) * 64 + ks * 32 + hi * 16);
            s0 = wmma_bf16(qf[ks], b0, s0);
            v16bf b1 = ldB32(Kp + (size_t)(kc + 16 + lo) * 64 + ks * 32 + hi * 16);
            s1 = wmma_bf16(qf[ks], b1, s1);
        }

        // ---- online softmax (rows live in 16-lane halves)
        float p0[8], p1[8], alpha[8];
        for (int v = 0; v < 8; ++v) {
            float x0 = s0[v] * scale;
            float x1 = s1[v] * scale;
            float r = fmaxf(x0, x1);
            for (int off = 1; off < 16; off <<= 1)
                r = fmaxf(r, __shfl_xor(r, off, 16));
            float mn = fmaxf(mrow[v], r);
            float al = __expf(mrow[v] - mn);
            float e0 = __expf(x0 - mn);
            float e1 = __expf(x1 - mn);
            float rs = e0 + e1;
            for (int off = 1; off < 16; off <<= 1)
                rs += __shfl_xor(rs, off, 16);
            lrow[v] = lrow[v] * al + rs;
            mrow[v] = mn;
            alpha[v] = al;
            p0[v] = e0; p1[v] = e1;
        }
        for (int nt = 0; nt < 4; ++nt)
            for (int v = 0; v < 8; ++v) o[nt][v] *= alpha[v];

        // ---- transpose P via LDS: C-layout -> A-layout (bf16)
        for (int v = 0; v < 8; ++v) {
            const int r = v + 8 * hi;
            Pw[r * 32 + lo]      = f2bf(p0[v]);
            Pw[r * 32 + 16 + lo] = f2bf(p1[v]);
        }
        asm volatile("s_wait_dscnt 0" ::: "memory");
        V16 pa;
        pa.h[0] = *(const u16x8*)&Pw[lo * 32 + hi * 16];
        pa.h[1] = *(const u16x8*)&Pw[lo * 32 + hi * 16 + 8];

        // ---- O += P @ V   (V pre-transposed: rows contiguous in keys)
        for (int nt = 0; nt < 4; ++nt) {
            v16bf bv = ldB32(Vp + (size_t)(nt * 16 + lo) * 2048 + kc + hi * 16);
            o[nt] = wmma_bf16(pa.v, bv, o[nt]);
        }
    }

    // ---- finalize: O / l, store fp32 merged-head layout [b][s][h*64+d]
    for (int nt = 0; nt < 4; ++nt) {
        for (int v = 0; v < 8; ++v) {
            const int s = q_base + v + 8 * hi;
            const int d = nt * 16 + lo;
            Out[((size_t)(b * 2048 + s)) * 1024 + h * 64 + d] = o[nt][v] / lrow[v];
        }
    }
}

// ---------------------------------------------------------------------------
// Stage 3: Abf = bf16(Attn0 + Attn1)
// ---------------------------------------------------------------------------
__global__ void sum_to_bf16_kernel(const float* __restrict__ A,
                                   const float* __restrict__ B,
                                   unsigned short* __restrict__ O, int n) {
    int i = blockIdx.x * blockDim.x + threadIdx.x;
    if (i < n) O[i] = f2bf(A[i] + B[i]);
}

// ---------------------------------------------------------------------------
// Stage 4: output projection. out[4096,1024] = Abf @ Wout   (fp32 out)
// ---------------------------------------------------------------------------
__global__ __launch_bounds__(256)
void out_gemm_kernel(const unsigned short* __restrict__ Abf,   // [4096][1024] bf16
                     const unsigned short* __restrict__ WT,    // [1024][1024] bf16 (N-major)
                     float* __restrict__ out) {
    const int lane = threadIdx.x & 31;
    const int wave = threadIdx.x >> 5;
    const int lo   = lane & 15;
    const int hi   = lane >> 4;

    const int m_tile = (blockIdx.y * 8 + wave) * 16;
    const int n_base = blockIdx.x * 128;

    v8f acc[8];
    for (int i = 0; i < 8; ++i)
        for (int j = 0; j < 8; ++j) acc[i][j] = 0.0f;

    const unsigned short* __restrict__ arow = Abf + (size_t)(m_tile + lo) * 1024;

    for (int k0 = 0; k0 < 1024; k0 += 32) {
        __builtin_prefetch(arow + k0 + 64, 0, 0);
        V16 a;
        a.h[0] = *(const u16x8*)(arow + k0 + hi * 16);
        a.h[1] = *(const u16x8*)(arow + k0 + hi * 16 + 8);

        for (int nt = 0; nt < 8; ++nt) {
            const int n = n_base + nt * 16 + lo;
            v16bf b = ldB32(WT + (size_t)n * 1024 + k0 + hi * 16);
            acc[nt] = wmma_bf16(a.v, b, acc[nt]);
        }
    }

    for (int nt = 0; nt < 8; ++nt) {
        const int n = n_base + nt * 16 + lo;
        for (int v = 0; v < 8; ++v) {
            const int m = m_tile + v + 8 * hi;
            out[(size_t)m * 1024 + n] = acc[nt][v];
        }
    }
}

// ---------------------------------------------------------------------------
// Launch
// ---------------------------------------------------------------------------
extern "C" void kernel_launch(void* const* d_in, const int* in_sizes, int n_in,
                              void* d_out, int out_size, void* d_ws, size_t ws_size,
                              hipStream_t stream) {
    const float* Xv   = (const float*)d_in[0];   // [2,2048,1024]
    const float* Xt   = (const float*)d_in[1];   // [2,2048,1024]
    const float* Wqkv = (const float*)d_in[2];   // [1024,3072]
    const float* Wout = (const float*)d_in[3];   // [1024,1024]
    float* out = (float*)d_out;                  // [2,2048,1024]

    char* ws = (char*)d_ws;
    // workspace layout (bytes)
    unsigned short* WqkvT = (unsigned short*)(ws + 0);                      //  6,291,456
    unsigned short* WoutT = (unsigned short*)(ws + 6291456);                //  2,097,152
    unsigned short* Qb    = (unsigned short*)(ws + 8388608);                // 16,777,216
    unsigned short* Kb    = (unsigned short*)(ws + 25165824);               // 16,777,216
    unsigned short* Vt    = (unsigned short*)(ws + 41943040);               // 16,777,216
    float*          Attn0 = (float*)(ws + 58720256);                        // 16,777,216
    float*          Attn1 = (float*)(ws + 75497472);                        // 16,777,216
    unsigned short* Abf   = (unsigned short*)(ws + 92274688);               //  8,388,608
    // total: 100,663,296 bytes

    // Stage 0: weight transposes -> bf16 N-major
    transpose_bf16_kernel<<<(3072 * 1024 + 255) / 256, 256, 0, stream>>>(Wqkv, WqkvT, 1024, 3072);
    transpose_bf16_kernel<<<(1024 * 1024 + 255) / 256, 256, 0, stream>>>(Wout, WoutT, 1024, 1024);

    // Stage 1: QKV projection (both modalities), scatter Q/K row-major, V transposed
    qkv_gemm_kernel<<<dim3(24, 32, 2), 256, 0, stream>>>(Xv, Xt, WqkvT, Qb, Kb, Vt);

    // Stage 2: bidirectional cross attention (flash style)
    attention_kernel<<<dim3(32, 32, 2), 128, 0, stream>>>(Qb, Kb, Vt, Attn0, Attn1);

    // Stage 3: sum directions -> bf16
    sum_to_bf16_kernel<<<(4096 * 1024 + 255) / 256, 256, 0, stream>>>(Attn0, Attn1, Abf, 4096 * 1024);

    // Stage 4: output projection -> fp32
    out_gemm_kernel<<<dim3(8, 32), 256, 0, stream>>>(Abf, WoutT, out);
}
